// CausalSelfAttention_68367289418387
// MI455X (gfx1250) — matrix operand order repaired and checked by reading, compiled, not run on hardware
//
#include <hip/hip_runtime.h>

// Problem constants (match reference)
#define BB 8
#define TT 2048
#define CC 768
#define HH 12
#define DD 64
// 3C = 2304; T is a power of two (shifts for row decomposition)

typedef __attribute__((ext_vector_type(16))) __bf16         bf16x16;
typedef __attribute__((ext_vector_type(8)))  float          f32x8;
typedef __attribute__((ext_vector_type(8)))  unsigned short ushortx8;

union BFrag { bf16x16 v; ushortx8 h[2]; };

// float -> bf16, round-to-nearest-even (inputs are finite; no NaN path needed)
__device__ __forceinline__ unsigned short f2bf(float f) {
  unsigned int u = __float_as_uint(f);
  u += 0x7FFFu + ((u >> 16) & 1u);
  return (unsigned short)(u >> 16);
}

// A-fragment (16x32, 16-bit): lane holds row `row`, K runs {kb..kb+7, kb+16..kb+23},
// kb = kstart + hi*8. Two 16B contiguous loads.
__device__ __forceinline__ bf16x16 load_a_frag(const unsigned short* base, int row,
                                               int ld, int kstart, int hi) {
  const unsigned short* p = base + (long)row * ld + kstart + (hi << 3);
  BFrag f;
  f.h[0] = *(const ushortx8*)(p);
  f.h[1] = *(const ushortx8*)(p + 16);
  return f.v;
}

// B-fragment (32x16, 16-bit) from K-major ("transposed") storage:
// K runs {kb..kb+7, kb+16..kb+23}, kb = kstart + hi*8, at base[nrow*ld + k].
__device__ __forceinline__ bf16x16 load_b_frag(const unsigned short* base, int nrow,
                                               int ld, int kstart, int hi) {
  const unsigned short* p = base + (long)nrow * ld + kstart + (hi << 3);
  BFrag f;
  f.h[0] = *(const ushortx8*)(p);
  f.h[1] = *(const ushortx8*)(p + 16);
  return f.v;
}

__device__ __forceinline__ f32x8 wmma_bf16(bf16x16 a, bf16x16 b, f32x8 c) {
  return __builtin_amdgcn_wmma_f32_16x16x32_bf16(false, a, false, b, (short)0, c,
                                                 false, false);
}

// ---------------- conversion / transpose kernels ----------------

__global__ void __launch_bounds__(256) k_cvt_x(const float* __restrict__ x,
                                               unsigned short* __restrict__ xb, int n) {
  int i = blockIdx.x * 256 + threadIdx.x;
  if (i < n) xb[i] = f2bf(x[i]);
}

// w: [rows=K][cols=N] f32 -> wt: [N][K] bf16
__global__ void __launch_bounds__(256) k_transpose_w(const float* __restrict__ w,
                                                     unsigned short* __restrict__ wt,
                                                     int rows, int cols) {
  int i = blockIdx.x * 256 + threadIdx.x;
  if (i < rows * cols) {
    int n = i / rows;
    int k = i % rows;
    wt[i] = f2bf(w[(long)k * cols + n]);
  }
}

// ---------------- QKV projection: qkv = x @ w_attn + b_attn ----------------
// One wave per 32x64 output tile of [B*T, 3C] (2x4 register blocking: 8 WMMAs
// per 32-deep k-step from 6 fragment loads). A 64-wide N tile never crosses a
// q/k/v section or head boundary, so section/head math is wave-uniform.

__global__ void __launch_bounds__(256) k_qkv(const unsigned short* __restrict__ xb,
                                             const unsigned short* __restrict__ wat,
                                             const float* __restrict__ b_attn,
                                             unsigned short* __restrict__ qb,
                                             unsigned short* __restrict__ kbuf,
                                             unsigned short* __restrict__ vt,
                                             float* __restrict__ k_out,
                                             float* __restrict__ v_out) {
  const int lane = threadIdx.x & 31;
  const int wave = threadIdx.x >> 5;
  const int tile = blockIdx.x * 8 + wave;
  const int NTN = (3 * CC) / 64;  // 36 column super-tiles
  const int tm = tile / NTN;
  const int tn = tile % NTN;
  const int lo = lane & 15, hi = lane >> 4;

  const int rowA0 = tm * 32 + lo;

  f32x8 acc[2][4] = {};
#pragma unroll 2
  for (int kc = 0; kc < CC; kc += 32) {
    bf16x16 a0 = load_a_frag(xb, rowA0, CC, kc, hi);
    bf16x16 a1 = load_a_frag(xb, rowA0 + 16, CC, kc, hi);
    bf16x16 b0 = load_b_frag(wat, tn * 64 + 0 + lo, CC, kc, hi);
    bf16x16 b1 = load_b_frag(wat, tn * 64 + 16 + lo, CC, kc, hi);
    bf16x16 b2 = load_b_frag(wat, tn * 64 + 32 + lo, CC, kc, hi);
    bf16x16 b3 = load_b_frag(wat, tn * 64 + 48 + lo, CC, kc, hi);
    acc[0][0] = wmma_bf16(a0, b0, acc[0][0]);
    acc[0][1] = wmma_bf16(a0, b1, acc[0][1]);
    acc[0][2] = wmma_bf16(a0, b2, acc[0][2]);
    acc[0][3] = wmma_bf16(a0, b3, acc[0][3]);
    acc[1][0] = wmma_bf16(a1, b0, acc[1][0]);
    acc[1][1] = wmma_bf16(a1, b1, acc[1][1]);
    acc[1][2] = wmma_bf16(a1, b2, acc[1][2]);
    acc[1][3] = wmma_bf16(a1, b3, acc[1][3]);
  }

  // wave-uniform section/head decomposition (tn*64 aligned to both)
  const int colbase = tn * 64;
  const int sect = colbase / CC;        // 0=q, 1=k, 2=v
  const int h = (colbase % CC) / DD;    // head index
  float bias[4];
#pragma unroll
  for (int j = 0; j < 4; ++j) bias[j] = b_attn[colbase + j * 16 + lo];

#pragma unroll
  for (int i = 0; i < 2; ++i) {
    const int rbase = tm * 32 + i * 16 + (hi << 3);
#pragma unroll
    for (int r = 0; r < 8; ++r) {
      const int row = rbase + r;        // global row in [0, B*T)
      const int b_ = row >> 11;         // /T (T=2048)
      const int t = row & (TT - 1);     // %T
      const long hbase = (((long)(b_ * HH + h)) * TT + t) * DD;
#pragma unroll
      for (int j = 0; j < 4; ++j) {
        const float v = acc[i][j][r] + bias[j];
        const int d = j * 16 + lo;
        const long idx = hbase + d;     // [B,H,T,D]
        if (sect == 0) {
          qb[idx] = f2bf(v);
        } else if (sect == 1) {
          kbuf[idx] = f2bf(v);
          k_out[idx] = v;
        } else {
          v_out[idx] = v;
          const long ti = (((long)(b_ * HH + h)) * DD + d) * TT + t;  // [B,H,D,T]
          vt[ti] = f2bf(v);
        }
      }
    }
  }
}

// ---------------- flash attention (causal, online softmax) ----------------
// One wave per (b, h, 16-row query tile); 32 keys per step; P via per-wave LDS.

__global__ void __launch_bounds__(256) k_attn(const unsigned short* __restrict__ qb,
                                              const unsigned short* __restrict__ kbuf,
                                              const unsigned short* __restrict__ vt,
                                              unsigned short* __restrict__ yb) {
  __shared__ __align__(16) unsigned short plds[8 * 16 * 32];  // 1KB per wave
  const int lane = threadIdx.x & 31;
  const int wave = threadIdx.x >> 5;
  unsigned short* pw = plds + wave * 512;

  const int w = blockIdx.x * 8 + wave;
  const int QT = TT / 16;
  const int qt = w % QT;
  const int hb = w / QT;
  const int h = hb % HH;
  const int b = hb / HH;
  const int lo = lane & 15, hi = lane >> 4;

  const unsigned short* qbase = qb + (((long)(b * HH + h)) * TT) * DD;
  const unsigned short* kbase = kbuf + (((long)(b * HH + h)) * TT) * DD;
  const unsigned short* vbase = vt + (((long)(b * HH + h)) * DD) * TT;

  const int qrow = qt * 16 + lo;
  const bf16x16 qf0 = load_a_frag(qbase, qrow, DD, 0, hi);   // d 0..31
  const bf16x16 qf1 = load_a_frag(qbase, qrow, DD, 32, hi);  // d 32..63

  f32x8 acc0 = {}, acc1 = {}, acc2 = {}, acc3 = {};
  float mrow[8], lrow[8];
#pragma unroll
  for (int r = 0; r < 8; ++r) { mrow[r] = -1e30f; lrow[r] = 0.f; }

  const int nsteps = (qt * 16 + 16 + 31) / 32;  // ceil((qt+1)*16 / 32)
  const float scale = 0.125f;                   // 1/sqrt(64)

  for (int s = 0; s < nsteps; ++s) {
    const int kb0 = s * 32;

    if (s + 1 < nsteps) {  // prefetch next step's K rows and V columns
      __builtin_prefetch(kbase + (long)(kb0 + 32 + lane) * DD, 0, 3);
      __builtin_prefetch(vbase + (long)lane * TT + kb0 + 32, 0, 3);
      __builtin_prefetch(vbase + (long)(32 + lane) * TT + kb0 + 32, 0, 3);
    }

    // S = Q @ K^T for 32 keys: two 16x16 f32 fragments
    f32x8 s0 = {}, s1 = {};
    s0 = wmma_bf16(qf0, load_b_frag(kbase, kb0 + lo, DD, 0, hi), s0);
    s0 = wmma_bf16(qf1, load_b_frag(kbase, kb0 + lo, DD, 32, hi), s0);
    s1 = wmma_bf16(qf0, load_b_frag(kbase, kb0 + 16 + lo, DD, 0, hi), s1);
    s1 = wmma_bf16(qf1, load_b_frag(kbase, kb0 + 16 + lo, DD, 32, hi), s1);

#pragma unroll
    for (int r = 0; r < 8; ++r) { s0[r] *= scale; s1[r] *= scale; }

    if (kb0 + 31 > qt * 16) {  // only diagonal region needs the causal mask
#pragma unroll
      for (int r = 0; r < 8; ++r) {
        int row = qt * 16 + r + (hi << 3);
        if (kb0 + lo > row) s0[r] = -1e30f;
        if (kb0 + 16 + lo > row) s1[r] = -1e30f;
      }
    }

    // online softmax: row stats live replicated across the 16-lane half-groups
#pragma unroll
    for (int r = 0; r < 8; ++r) {
      float mx = fmaxf(s0[r], s1[r]);
      mx = fmaxf(mx, __shfl_xor(mx, 1));
      mx = fmaxf(mx, __shfl_xor(mx, 2));
      mx = fmaxf(mx, __shfl_xor(mx, 4));
      mx = fmaxf(mx, __shfl_xor(mx, 8));
      float mn = fmaxf(mrow[r], mx);
      float corr = __expf(mrow[r] - mn);
      float p0 = __expf(s0[r] - mn);
      float p1 = __expf(s1[r] - mn);
      float rs = p0 + p1;
      rs += __shfl_xor(rs, 1);
      rs += __shfl_xor(rs, 2);
      rs += __shfl_xor(rs, 4);
      rs += __shfl_xor(rs, 8);
      lrow[r] = lrow[r] * corr + rs;
      mrow[r] = mn;
      acc0[r] *= corr; acc1[r] *= corr; acc2[r] *= corr; acc3[r] *= corr;
      // stash P (C-layout) into LDS row-major [16 x 32]
      pw[(r + (hi << 3)) * 32 + lo] = f2bf(p0);
      pw[(r + (hi << 3)) * 32 + lo + 16] = f2bf(p1);
    }
    asm volatile("s_wait_dscnt 0" ::: "memory");  // P visible before A-frag reload

    // reload P in A-fragment layout (two ds_load_b128 per lane) and do P @ V
    bf16x16 pf = load_a_frag(pw, lo, 32, 0, hi);
    acc0 = wmma_bf16(pf, load_b_frag(vbase, 0 * 16 + lo, TT, kb0, hi), acc0);
    acc1 = wmma_bf16(pf, load_b_frag(vbase, 1 * 16 + lo, TT, kb0, hi), acc1);
    acc2 = wmma_bf16(pf, load_b_frag(vbase, 2 * 16 + lo, TT, kb0, hi), acc2);
    acc3 = wmma_bf16(pf, load_b_frag(vbase, 3 * 16 + lo, TT, kb0, hi), acc3);
  }

  // normalize and emit bf16 y_att in [B,T,C] layout for the projection GEMM
#pragma unroll
  for (int r = 0; r < 8; ++r) {
    float inv = 1.0f / lrow[r];
    int row = qt * 16 + r + (hi << 3);
    long obase = ((long)b * TT + row) * CC + h * DD;
    yb[obase + 0 + lo] = f2bf(acc0[r] * inv);
    yb[obase + 16 + lo] = f2bf(acc1[r] * inv);
    yb[obase + 32 + lo] = f2bf(acc2[r] * inv);
    yb[obase + 48 + lo] = f2bf(acc3[r] * inv);
  }
}

// ---------------- output projection: y = y_att @ w_proj + b_proj ----------------
// Same 2x4 register blocking as k_qkv.

__global__ void __launch_bounds__(256) k_proj(const unsigned short* __restrict__ yb,
                                              const unsigned short* __restrict__ wpt,
                                              const float* __restrict__ b_proj,
                                              float* __restrict__ out) {
  const int lane = threadIdx.x & 31;
  const int wave = threadIdx.x >> 5;
  const int tile = blockIdx.x * 8 + wave;
  const int NTN = CC / 64;  // 12 column super-tiles
  const int tm = tile / NTN;
  const int tn = tile % NTN;
  const int lo = lane & 15, hi = lane >> 4;

  const int rowA0 = tm * 32 + lo;

  f32x8 acc[2][4] = {};
#pragma unroll 2
  for (int kc = 0; kc < CC; kc += 32) {
    bf16x16 a0 = load_a_frag(yb, rowA0, CC, kc, hi);
    bf16x16 a1 = load_a_frag(yb, rowA0 + 16, CC, kc, hi);
    bf16x16 b0 = load_b_frag(wpt, tn * 64 + 0 + lo, CC, kc, hi);
    bf16x16 b1 = load_b_frag(wpt, tn * 64 + 16 + lo, CC, kc, hi);
    bf16x16 b2 = load_b_frag(wpt, tn * 64 + 32 + lo, CC, kc, hi);
    bf16x16 b3 = load_b_frag(wpt, tn * 64 + 48 + lo, CC, kc, hi);
    acc[0][0] = wmma_bf16(a0, b0, acc[0][0]);
    acc[0][1] = wmma_bf16(a0, b1, acc[0][1]);
    acc[0][2] = wmma_bf16(a0, b2, acc[0][2]);
    acc[0][3] = wmma_bf16(a0, b3, acc[0][3]);
    acc[1][0] = wmma_bf16(a1, b0, acc[1][0]);
    acc[1][1] = wmma_bf16(a1, b1, acc[1][1]);
    acc[1][2] = wmma_bf16(a1, b2, acc[1][2]);
    acc[1][3] = wmma_bf16(a1, b3, acc[1][3]);
  }

  float bias[4];
#pragma unroll
  for (int j = 0; j < 4; ++j) bias[j] = b_proj[tn * 64 + j * 16 + lo];

#pragma unroll
  for (int i = 0; i < 2; ++i) {
    const int rbase = tm * 32 + i * 16 + (hi << 3);
#pragma unroll
    for (int r = 0; r < 8; ++r) {
      const long rowoff = (long)(rbase + r) * CC + tn * 64 + lo;
#pragma unroll
      for (int j = 0; j < 4; ++j) {
        out[rowoff + j * 16] = acc[i][j][r] + bias[j];
      }
    }
  }
}

// ---------------- host-side launch ----------------

extern "C" void kernel_launch(void* const* d_in, const int* in_sizes, int n_in,
                              void* d_out, int out_size, void* d_ws, size_t ws_size,
                              hipStream_t stream) {
  (void)in_sizes; (void)n_in; (void)out_size; (void)ws_size;
  const float* x      = (const float*)d_in[0];
  const float* w_attn = (const float*)d_in[1];
  const float* b_attn = (const float*)d_in[2];
  const float* w_proj = (const float*)d_in[3];
  const float* b_proj = (const float*)d_in[4];

  float* y_out = (float*)d_out;                       // [B,T,C]
  float* k_out = y_out + (size_t)BB * TT * CC;        // [B,H,T,D]
  float* v_out = k_out + (size_t)BB * TT * CC;        // [B,H,T,D]

  char* ws = (char*)d_ws;
  size_t off = 0;
  auto alloc = [&](size_t bytes) -> void* {
    void* p = ws + off;
    off += (bytes + 255) & ~(size_t)255;
    return p;
  };
  const size_t NBTC = (size_t)BB * TT * CC;
  unsigned short* xb  = (unsigned short*)alloc(NBTC * 2);
  unsigned short* wat = (unsigned short*)alloc((size_t)3 * CC * CC * 2);
  unsigned short* wpt = (unsigned short*)alloc((size_t)CC * CC * 2);
  unsigned short* qb  = (unsigned short*)alloc(NBTC * 2);
  unsigned short* kb  = (unsigned short*)alloc(NBTC * 2);
  unsigned short* vt  = (unsigned short*)alloc(NBTC * 2);
  unsigned short* yb  = (unsigned short*)alloc(NBTC * 2);

  const int nx = BB * TT * CC;
  k_cvt_x<<<(nx + 255) / 256, 256, 0, stream>>>(x, xb, nx);
  k_transpose_w<<<(3 * CC * CC + 255) / 256, 256, 0, stream>>>(w_attn, wat, CC, 3 * CC);
  k_transpose_w<<<(CC * CC + 255) / 256, 256, 0, stream>>>(w_proj, wpt, CC, CC);

  const int tiles1 = (BB * TT / 32) * (3 * CC / 64);   // 18432 wave-tiles
  k_qkv<<<tiles1 / 8, 256, 0, stream>>>(xb, wat, b_attn, qb, kb, vt, k_out, v_out);

  const int wtiles = BB * HH * (TT / 16);              // 12288 query tiles
  k_attn<<<wtiles / 8, 256, 0, stream>>>(qb, kb, vt, yb);

  const int tiles3 = (BB * TT / 32) * (CC / 64);       // 6144 wave-tiles
  k_proj<<<tiles3 / 8, 256, 0, stream>>>(yb, wpt, b_proj, y_out);
}
